// Shift_Soft_75350906241312
// MI455X (gfx1250) — matrix operand order repaired and checked by reading, not compiled, not run on hardware
//
#include <hip/hip_runtime.h>
#include <hip/hip_bf16.h>
#include <math.h>

typedef __attribute__((ext_vector_type(16))) __bf16 v16bf;
typedef __attribute__((ext_vector_type(8)))  __bf16 v8bf;
typedef __attribute__((ext_vector_type(8)))  float  v8f;

// Problem constants (reference: B=2, C=128, H=W=128, rate=2)
#define NB   2
#define CIN  128
#define CF   64
#define HW   128
#define HS   64
#define LQ   4096            // 64*64
#define KDIM 576             // 64 * 3 * 3

// ---------------- workspace layout (bytes) ----------------
static constexpr size_t OFF_FP = 0;                                       // [NB][LQ][KDIM] bf16
static constexpr size_t SZ_FP  = (size_t)NB * LQ * KDIM * 2;
static constexpr size_t OFF_WN = OFF_FP + SZ_FP;                          // [NB][LQ][KDIM] bf16
static constexpr size_t OFF_MM = OFF_WN + SZ_FP;                          // [LQ] f32
static constexpr size_t OFF_YI = OFF_MM + (size_t)LQ * 4;                 // [NB][LQ][LQ] f32
static constexpr size_t SZ_YI  = (size_t)NB * LQ * LQ * 4;
static constexpr size_t OFF_FO = OFF_YI + SZ_YI;                          // [NB][LQ][LQ] f32
static constexpr size_t OFF_PT = OFF_FO + SZ_YI;                          // [NB][LQ][LQ] bf16 (transposed)
static constexpr size_t OFF_AT = OFF_PT + (size_t)NB * LQ * LQ * 2;       // [NB][16][CF][LQ] bf16

// ---------------------------------------------------------------------------
// 1) Build Fp (raw 3x3xC patches of downsampled f) and Wn (L2-normalized),
//    both bf16, K-major rows of 576.  One block (64 threads = channels) per q.
// ---------------------------------------------------------------------------
__global__ void k_prep_fw(const float* __restrict__ in, __bf16* __restrict__ Fp,
                          __bf16* __restrict__ Wn) {
  const int q = blockIdx.x;            // 0..4095
  const int b = blockIdx.y;            // batch
  const int c = threadIdx.x;           // 0..63
  const int qy = q >> 6, qx = q & 63;
  const float* src = in + ((size_t)b * CIN + c) * (HW * HW);
  float v[9];
  float s = 0.f;
  #pragma unroll
  for (int ky = 0; ky < 3; ++ky) {
    #pragma unroll
    for (int kx = 0; kx < 3; ++kx) {
      int y = qy + ky - 1, x = qx + kx - 1;
      float val = 0.f;
      if ((unsigned)y < (unsigned)HS && (unsigned)x < (unsigned)HS)
        val = src[(2 * y) * HW + 2 * x];          // downsample ::2
      v[ky * 3 + kx] = val;
      s += val * val;
    }
  }
  __shared__ float red[64];
  red[c] = s;
  __syncthreads();
  for (int off = 32; off > 0; off >>= 1) {
    if (c < off) red[c] += red[c + off];
    __syncthreads();
  }
  const float rnorm = 1.f / fmaxf(sqrtf(red[0]), 1e-8f);
  __bf16* fr = Fp + ((size_t)b * LQ + q) * KDIM + c * 9;
  __bf16* wr = Wn + ((size_t)b * LQ + q) * KDIM + c * 9;
  #pragma unroll
  for (int k = 0; k < 9; ++k) {
    fr[k] = (__bf16)v[k];
    wr[k] = (__bf16)(v[k] * rnorm);
  }
}

// ---------------------------------------------------------------------------
// 2) mm[q] = 1 if the 3x3 neighborhood (zero-padded) of mask is all zero.
// ---------------------------------------------------------------------------
__global__ void k_mask_mm(const int* __restrict__ mask, float* __restrict__ mm) {
  int q = blockIdx.x * blockDim.x + threadIdx.x;
  if (q >= LQ) return;
  int qy = q >> 6, qx = q & 63;
  int any = 0;
  #pragma unroll
  for (int ky = -1; ky <= 1; ++ky)
    #pragma unroll
    for (int kx = -1; kx <= 1; ++kx) {
      int y = qy + ky, x = qx + kx;
      if ((unsigned)y < (unsigned)HS && (unsigned)x < (unsigned)HS)
        any |= mask[y * HS + x];
    }
  mm[q] = any ? 0.f : 1.f;
}

// ---------------------------------------------------------------------------
// 3) Similarity GEMM: yi[q,p] = sum_k Wn[q,k] * Fp[p,k]   (bf16 x bf16 -> f32)
//    Register-blocked: each wave computes a 16(q) x 32(p) tile -> the A
//    fragment is reused by two B fragments, and two independent WMMAs per
//    K-chunk hide the bf16 WMMA RAW latency.  8 waves -> 64q x 64p per WG.
// ---------------------------------------------------------------------------
__global__ void __launch_bounds__(256) k_gemm_sim(const __bf16* __restrict__ Wn,
                                                  const __bf16* __restrict__ Fp,
                                                  float* __restrict__ yi) {
  const int lane = threadIdx.x & 31;
  const int w    = threadIdx.x >> 5;
  const int b    = blockIdx.z;
  const int q0   = blockIdx.x * 64 + (w & 3) * 16;
  const int p0   = blockIdx.y * 64 + (w >> 2) * 32;   // wave covers p0 .. p0+31
  const int row  = lane & 15;
  const int half = lane >> 4;
  const __bf16* arow  = Wn + ((size_t)b * LQ + q0 + row) * KDIM;  // A: M=q, K contiguous
  const __bf16* brow0 = Fp + ((size_t)b * LQ + p0 + row) * KDIM;  // B: N=p, K contiguous
  const __bf16* brow1 = brow0 + (size_t)16 * KDIM;
  const int aoff = half * 16;   // A lanes<16: K0-15; lanes>=16: K16-31
  const int boff = half * 16;   // B lanes<16: K0-15; lanes>=16: K16-31

  v8f acc0 = {}, acc1 = {};
  #pragma unroll 2
  for (int kb = 0; kb < KDIM; kb += 32) {
    union { v16bf v; v8bf h[2]; } a;
    a.h[0] = *(const v8bf*)(arow + kb + aoff);
    a.h[1] = *(const v8bf*)(arow + kb + 8 + aoff);
    v16bf b0 = *(const v16bf*)(brow0 + kb + boff);
    v16bf b1 = *(const v16bf*)(brow1 + kb + boff);
    acc0 = __builtin_amdgcn_wmma_f32_16x16x32_bf16(false, a.v, false, b0,
                                                   (short)0, acc0, false, false);
    acc1 = __builtin_amdgcn_wmma_f32_16x16x32_bf16(false, a.v, false, b1,
                                                   (short)0, acc1, false, false);
  }
  const int N = lane & 15;
  float* out = yi + ((size_t)b * LQ + q0) * (size_t)LQ + p0 + N;
  #pragma unroll
  for (int r = 0; r < 8; ++r) {
    int M = (lane < 16) ? r : (8 + r);
    out[(size_t)M * LQ] = acc0[r];
    out[(size_t)M * LQ + 16] = acc1[r];
  }
}

// ---------------------------------------------------------------------------
// 4) Fused double "fuse conv" (3x3 identity kernel, transpose sandwich):
//    F[q,p] = sum_{s,t in [-1,1]} yi[sw(sw(q)+s)+t, sw(sw(p)+s)+t], zero-padded.
//    sw(i) swaps the (y,x) sub-indices of the flattened 64x64 index.
//    One 64 MB batch slice fits in the 192 MB L2, so the 9 reads are L2-served.
// ---------------------------------------------------------------------------
__global__ void k_fuse(const float* __restrict__ yi, float* __restrict__ Fo) {
  const int p = blockIdx.x * 256 + threadIdx.x;
  const int q = blockIdx.y;
  const int b = blockIdx.z;
  const float* src = yi + (size_t)b * LQ * LQ;
  const int swq = ((q & 63) << 6) | (q >> 6);
  const int swp = ((p & 63) << 6) | (p >> 6);
  float acc = 0.f;
  #pragma unroll
  for (int s = -1; s <= 1; ++s) {
    int iq = swq + s, ip = swp + s;
    if ((unsigned)iq >= (unsigned)LQ || (unsigned)ip >= (unsigned)LQ) continue;
    int aq = ((iq & 63) << 6) | (iq >> 6);
    int ap = ((ip & 63) << 6) | (ip >> 6);
    #pragma unroll
    for (int t = -1; t <= 1; ++t) {
      int rq = aq + t, rp = ap + t;
      if ((unsigned)rq < (unsigned)LQ && (unsigned)rp < (unsigned)LQ)
        acc += src[(size_t)rq * LQ + rp];
    }
  }
  Fo[(size_t)b * LQ * LQ + (size_t)q * LQ + p] = acc;
}

// ---------------------------------------------------------------------------
// 5) Masked softmax over q (in place). Block: 64 columns x 4 q-slices,
//    online max/sum + LDS merge.  x = F*mm*10; out = exp(x-M)/S * mm.
// ---------------------------------------------------------------------------
__global__ void __launch_bounds__(256) k_softmax(float* __restrict__ Fo,
                                                 const float* __restrict__ mm) {
  const int col = blockIdx.x * 64 + (threadIdx.x & 63);
  const int sub = threadIdx.x >> 6;
  const int b   = blockIdx.z;
  float* base = Fo + (size_t)b * LQ * LQ + col;
  float m = -1e30f, s = 0.f;
  for (int q = sub; q < LQ; q += 4) {
    float x = base[(size_t)q * LQ] * mm[q] * 10.f;
    float mn = fmaxf(m, x);
    s = s * expf(m - mn) + expf(x - mn);
    m = mn;
  }
  __shared__ float sm[256], ss[256];
  sm[threadIdx.x] = m; ss[threadIdx.x] = s;
  __syncthreads();
  if (sub == 0) {
    int c = threadIdx.x;
    float M = sm[c], S = ss[c];
    #pragma unroll
    for (int k = 1; k < 4; ++k) {
      float m2 = sm[c + 64 * k], s2 = ss[c + 64 * k];
      float mn = fmaxf(M, m2);
      S = S * expf(M - mn) + s2 * expf(m2 - mn);
      M = mn;
    }
    sm[c] = M; ss[c] = S;
  }
  __syncthreads();
  const float M    = sm[threadIdx.x & 63];
  const float Sinv = 1.f / ss[threadIdx.x & 63];
  for (int q = sub; q < LQ; q += 4) {
    float x = base[(size_t)q * LQ] * mm[q] * 10.f;
    base[(size_t)q * LQ] = expf(x - M) * Sinv * mm[q];
  }
}

// ---------------------------------------------------------------------------
// 6) Transpose P[L][p] f32 -> Pt[p][L] bf16 via LDS 32x32 tile.
// ---------------------------------------------------------------------------
__global__ void k_transpose(const float* __restrict__ P, __bf16* __restrict__ Pt) {
  __shared__ float tile[32][33];
  const int b  = blockIdx.z;
  const int q0 = blockIdx.x * 32, p0 = blockIdx.y * 32;
  const int tx = threadIdx.x, ty = threadIdx.y;   // 32 x 8
  const float* src = P + (size_t)b * LQ * LQ;
  #pragma unroll
  for (int i = 0; i < 4; ++i) {
    int r = ty + i * 8;
    tile[r][tx] = src[(size_t)(q0 + r) * LQ + p0 + tx];
  }
  __syncthreads();
  __bf16* dst = Pt + (size_t)b * LQ * LQ;
  #pragma unroll
  for (int i = 0; i < 4; ++i) {
    int r = ty + i * 8;
    dst[(size_t)(p0 + r) * LQ + q0 + tx] = (__bf16)tile[tx][r];
  }
}

// ---------------------------------------------------------------------------
// 7) Materialize raw_w as 16 tap planes At[b][kh*4+kw][c][L] bf16 (L-major).
//    raw_w[L,c,kh,kw] = in[b,c, 2qy+kh-1, 2qx+kw-1] (zero-padded).
// ---------------------------------------------------------------------------
__global__ void k_prep_raww(const float* __restrict__ in, __bf16* __restrict__ At) {
  const int l  = blockIdx.x * 256 + threadIdx.x;   // 0..4095
  const int tc = blockIdx.y;                       // tap*64 + c
  const int b  = blockIdx.z;
  const int tap = tc >> 6, c = tc & 63;
  const int kh = tap >> 2, kw = tap & 3;
  const int qy = l >> 6, qx = l & 63;
  const int iy = 2 * qy + kh - 1, ix = 2 * qx + kw - 1;
  float val = 0.f;
  if ((unsigned)iy < (unsigned)HW && (unsigned)ix < (unsigned)HW)
    val = in[(((size_t)b * CIN + c) * HW + iy) * HW + ix];
  At[(((size_t)b * 16 + tap) * CF + c) * LQ + l] = (__bf16)val;
}

// ---------------------------------------------------------------------------
// 8) Aggregation (transposed conv) as parity-decomposed WMMA GEMM:
//    For parity (r,s): O[c,m,n] = sum_{a,bb} sum_L At[tap][c][L] * Pt[j'][L]
//    with tap=(3-(r+2a))*4+(3-(s+2bb)), j'=(m+a-1+r)*64+(n+bb-1+s).
//    Writes out[b][128+c][2m+r][2n+s] = O/4 * mask[m,n].
//    Each lane streams 8 separate rows 8 KB apart at 64 B/chunk; a software
//    prefetch 1 KB ahead (every 8th chunk) keeps HBM latency hidden.
// ---------------------------------------------------------------------------
__global__ void __launch_bounds__(256) k_agg(const __bf16* __restrict__ At,
                                             const __bf16* __restrict__ Pt,
                                             const int* __restrict__ mask,
                                             float* __restrict__ out) {
  const int lane = threadIdx.x & 31;
  const int w    = threadIdx.x >> 5;
  const int b    = blockIdx.z;
  const int r    = blockIdx.y >> 1, s = blockIdx.y & 1;
  const int ctile = (w & 3) * 16;
  const int j0    = blockIdx.x * 32 + (w >> 2) * 16;
  const int nlane = lane & 15;
  const int half  = lane >> 4;
  const int j = j0 + nlane;
  const int m = j >> 6, nn = j & 63;

  const __bf16* browp[4];
  const __bf16* arowp[4];
  #pragma unroll
  for (int t = 0; t < 4; ++t) {
    const int a = t >> 1, bb = t & 1;
    const int iy = m + a - 1 + r;
    const int ix = nn + bb - 1 + s;
    const bool ok = ((unsigned)iy < (unsigned)HS) && ((unsigned)ix < (unsigned)HS);
    browp[t] = ok ? (Pt + ((size_t)b * LQ + iy * HS + ix) * LQ + half * 16) : nullptr;
    const int kh = r + 2 * a, kw = s + 2 * bb;
    const int tap = (3 - kh) * 4 + (3 - kw);
    arowp[t] = At + (((size_t)b * 16 + tap) * CF + ctile + (lane & 15)) * LQ + half * 16;
  }

  v8f acc = {};
  for (int kb = 0; kb < LQ; kb += 32) {
    if ((kb & 255) == 0) {               // uniform branch: prefetch 1KB ahead
      #pragma unroll
      for (int t = 0; t < 4; ++t) {
        if (browp[t]) __builtin_prefetch(browp[t] + kb + 512, 0, 1);
        __builtin_prefetch(arowp[t] + kb + 512, 0, 1);
      }
    }
    #pragma unroll
    for (int t = 0; t < 4; ++t) {
      union { v16bf v; v8bf h[2]; } a;
      a.h[0] = *(const v8bf*)(arowp[t] + kb);
      a.h[1] = *(const v8bf*)(arowp[t] + kb + 8);
      v16bf bm = {};
      if (browp[t]) bm = *(const v16bf*)(browp[t] + kb);
      acc = __builtin_amdgcn_wmma_f32_16x16x32_bf16(false, a.v, false, bm,
                                                    (short)0, acc, false, false);
    }
  }
  const float scale = 0.25f * (float)mask[m * HS + nn];
  const int oy = 2 * m + r, ox = 2 * nn + s;
  #pragma unroll
  for (int rr = 0; rr < 8; ++rr) {
    int c = ctile + ((lane < 16) ? rr : (8 + rr));
    out[(((size_t)b * 192 + 128 + c) * HW + oy) * HW + ox] = acc[rr] * scale;
  }
}

// ---------------------------------------------------------------------------
// 9) Output channels 0..127 = in_data[:, :64] duplicated (b_full, f_full).
// ---------------------------------------------------------------------------
__global__ void k_copy(const float* __restrict__ in, float* __restrict__ out) {
  size_t i = (size_t)blockIdx.x * 256 + threadIdx.x;
  const size_t total = (size_t)NB * 128 * HW * HW;
  if (i >= total) return;
  size_t hw = i & (HW * HW - 1);
  size_t c  = (i >> 14) & 127;
  size_t b  = i >> 21;
  out[((b * 192) + c) * (HW * HW) + hw] =
      in[((b * CIN) + (c & 63)) * (HW * HW) + hw];
}

// ---------------------------------------------------------------------------
extern "C" void kernel_launch(void* const* d_in, const int* in_sizes, int n_in,
                              void* d_out, int out_size, void* d_ws, size_t ws_size,
                              hipStream_t stream) {
  const float* in   = (const float*)d_in[0];
  const int*   mask = (const int*)d_in[1];
  float*       out  = (float*)d_out;
  char*        ws   = (char*)d_ws;

  __bf16* Fp = (__bf16*)(ws + OFF_FP);
  __bf16* Wn = (__bf16*)(ws + OFF_WN);
  float*  mm = (float*)(ws + OFF_MM);
  float*  yi = (float*)(ws + OFF_YI);
  float*  Fo = (float*)(ws + OFF_FO);
  __bf16* Pt = (__bf16*)(ws + OFF_PT);
  __bf16* At = (__bf16*)(ws + OFF_AT);

  k_prep_fw  <<<dim3(LQ, NB),        64,            0, stream>>>(in, Fp, Wn);
  k_mask_mm  <<<dim3(LQ / 256),      256,           0, stream>>>(mask, mm);
  k_prep_raww<<<dim3(16, 1024, NB),  256,           0, stream>>>(in, At);
  k_gemm_sim <<<dim3(64, 64, NB),    256,           0, stream>>>(Wn, Fp, yi);
  k_fuse     <<<dim3(16, LQ, NB),    256,           0, stream>>>(yi, Fo);
  k_softmax  <<<dim3(64, 1, NB),     256,           0, stream>>>(Fo, mm);
  k_transpose<<<dim3(128, 128, NB),  dim3(32, 8),   0, stream>>>(Fo, Pt);
  k_agg      <<<dim3(128, 4, NB),    256,           0, stream>>>(At, Pt, mask, out);
  k_copy     <<<dim3(16384),         256,           0, stream>>>(in, out);
}
